// MemoryBlock_28097676051170
// MI455X (gfx1250) — matrix operand order repaired and checked by reading, not compiled, not run on hardware
//
#include <hip/hip_runtime.h>
#include <math.h>

// ---------------------------------------------------------------------------
// Problem constants (from the reference)
// ---------------------------------------------------------------------------
constexpr int B_ = 8, L_ = 4096, G_ = 512, D_ = 512, H_ = 8, K_ = 64, DH_ = 64;
constexpr float SCALE_ = 0.125f;          // (D/H)^-0.5 = 64^-0.5
constexpr int BL_ = B_ * L_;              // 32768 tokens

typedef __attribute__((ext_vector_type(16))) _Float16 v16h;
typedef __attribute__((ext_vector_type(8)))  _Float16 v8h;
typedef __attribute__((ext_vector_type(8)))  float    v8f;
typedef __attribute__((ext_vector_type(4)))  float    v4f;

// ---------------------------------------------------------------------------
// Workspace layout (bytes)
// ---------------------------------------------------------------------------
constexpr size_t OFF_XH = 0;                                   // log1p(relu(x)) f16 [B*L, D]     32 MB
constexpr size_t OFF_KH = OFF_XH + (size_t)BL_ * D_ * 2;       // K f16 [B,H,L,DH]                32 MB
constexpr size_t OFF_VH = OFF_KH + (size_t)BL_ * D_ * 2;       // V f16 [B,H,L,DH]                32 MB
constexpr size_t OFF_QH = OFF_VH + (size_t)BL_ * D_ * 2;       // Q f16 [H,G,DH]                 512 KB
constexpr size_t OFF_WQ = OFF_QH + (size_t)H_ * G_ * DH_ * 2;  // Wq f16 [D,D]
constexpr size_t OFF_WK = OFF_WQ + (size_t)D_ * D_ * 2;
constexpr size_t OFF_WV = OFF_WK + (size_t)D_ * D_ * 2;
constexpr size_t OFF_WO = OFF_WV + (size_t)D_ * D_ * 2;
constexpr size_t OFF_TT = OFF_WO + (size_t)D_ * D_ * 2;        // target_token f16 [G,D]
constexpr size_t OFF_XO = OFF_TT + (size_t)G_ * D_ * 2;        // pooled f32 [B,H,G,DH]            8 MB
constexpr size_t OFF_AH = OFF_XO + (size_t)B_ * H_ * G_ * DH_ * 4; // normalized f16 [B*G, D]      4 MB
constexpr size_t OFF_MM = OFF_AH + (size_t)B_ * G_ * D_ * 2;   // encoded global min/max (2 u32)

// ---------------------------------------------------------------------------
// WMMA fragment loaders (CDNA5 16x16x32 f16 layouts, CDNA5 ISA)
// A 16x32 (MxK): lane 0-15 row M=lane holds K = koff..koff+7 and 16+koff..;
//                lanes 16-31 same rows, koff = 8.
// B 32x16 (KxN): built from N-major weight W[N,K] -> per-lane K = koff..koff+7
//                and 16+koff.. of one output column (same k order as A).
// ---------------------------------------------------------------------------
__device__ inline v16h load_a_frag(const _Float16* __restrict__ A, int lda,
                                   int m0, int k0, int lane) {
  int row  = m0 + (lane & 15);
  int koff = ((lane >> 4) & 1) * 8;
  const _Float16* p = A + (size_t)row * lda + k0 + koff;
  union { v16h v; v8h h[2]; } u;
  u.h[0] = *(const v8h*)(p);        // K = k0+koff .. +7
  u.h[1] = *(const v8h*)(p + 16);   // K = k0+16+koff .. +7
  return u.v;
}

__device__ inline v16h load_bt_frag(const _Float16* __restrict__ W, int ldw,
                                    int n0, int k0, int lane) {
  int col = n0 + (lane & 15);
  int kb  = k0 + ((lane >> 4) & 1) * 8;
  const _Float16* p = W + (size_t)col * ldw + kb;
  union { v16h v; v8h h[2]; } u;
  u.h[0] = *(const v8h*)(p);
  u.h[1] = *(const v8h*)(p + 16);
  return u.v;
}

__device__ inline v8f wmma_f16(v16h a, v16h b, v8f c) {
  return __builtin_amdgcn_wmma_f32_16x16x32_f16(false, a, false, b,
                                                (short)0, c, false, false);
}

// order-preserving float<->uint mapping for atomic global min/max
__device__ inline unsigned enc_f32(float f) {
  unsigned u = __float_as_uint(f);
  return (u & 0x80000000u) ? ~u : (u | 0x80000000u);
}
__device__ inline float dec_f32(unsigned u) {
  unsigned b = (u & 0x80000000u) ? (u & 0x7FFFFFFFu) : ~u;
  return __uint_as_float(b);
}

// ---------------------------------------------------------------------------
// Elementwise prep kernels
// ---------------------------------------------------------------------------
__global__ void cvt_f16_kernel(const float* __restrict__ src,
                               _Float16* __restrict__ dst, int n4) {
  int i = blockIdx.x * blockDim.x + threadIdx.x;
  if (i < n4) {
    v4f v = *(const v4f*)(src + 4 * i);
    _Float16* d = dst + 4 * i;
    d[0] = (_Float16)v.x; d[1] = (_Float16)v.y;
    d[2] = (_Float16)v.z; d[3] = (_Float16)v.w;
  }
}

__global__ void log1p_f16_kernel(const float* __restrict__ src,
                                 _Float16* __restrict__ dst, int n4) {
  int i = blockIdx.x * blockDim.x + threadIdx.x;
  if (i < n4) {
    v4f v = *(const v4f*)(src + 4 * i);
    _Float16* d = dst + 4 * i;
    d[0] = (_Float16)(v.x > 0.f ? log1pf(v.x) : 0.f);
    d[1] = (_Float16)(v.y > 0.f ? log1pf(v.y) : 0.f);
    d[2] = (_Float16)(v.z > 0.f ? log1pf(v.z) : 0.f);
    d[3] = (_Float16)(v.w > 0.f ? log1pf(v.w) : 0.f);
  }
}

__global__ void init_minmax_kernel(unsigned* __restrict__ mm) {
  mm[0] = 0xFFFFFFFFu;  // running min (encoded)
  mm[1] = 0x00000000u;  // running max (encoded)
}

// ---------------------------------------------------------------------------
// Fused K/V projection GEMM:  K,V[t,n] = sum_k xh[t,k] * W{k,v}[n,k] + b[n]
// Block = 256 threads (8 waves), tile 128(M) x 64(N); wave tile 32x32.
// A-fragment loaded once per k-step feeds BOTH outputs (8 WMMAs/k-step).
// Output stored head-major f16 [B,H,L,DH] so attention B-frags are contiguous.
// ---------------------------------------------------------------------------
__global__ void __launch_bounds__(256)
kv_gemm_kernel(const _Float16* __restrict__ xh, const _Float16* __restrict__ wk,
               const _Float16* __restrict__ wv, const float* __restrict__ bk,
               const float* __restrict__ bv, _Float16* __restrict__ kh,
               _Float16* __restrict__ vh) {
  const int lane = threadIdx.x & 31;
  const int wave = threadIdx.x >> 5;
  const int m0 = blockIdx.y * 128 + (wave >> 1) * 32;
  const int n0 = blockIdx.x * 64  + (wave & 1) * 32;

  v8f aK[2][2] = {};
  v8f aV[2][2] = {};
  for (int k0 = 0; k0 < D_; k0 += 32) {
    v16h a0 = load_a_frag(xh, D_, m0,      k0, lane);
    v16h a1 = load_a_frag(xh, D_, m0 + 16, k0, lane);
    v16h kb0 = load_bt_frag(wk, D_, n0,      k0, lane);
    v16h kb1 = load_bt_frag(wk, D_, n0 + 16, k0, lane);
    v16h vb0 = load_bt_frag(wv, D_, n0,      k0, lane);
    v16h vb1 = load_bt_frag(wv, D_, n0 + 16, k0, lane);
    // prefetch next A rows into cache (global_prefetch_b8)
    if (k0 + 32 < D_)
      __builtin_prefetch(xh + (size_t)(m0 + (lane & 15)) * D_ + k0 + 32, 0, 1);
    aK[0][0] = wmma_f16(a0, kb0, aK[0][0]);
    aK[0][1] = wmma_f16(a0, kb1, aK[0][1]);
    aK[1][0] = wmma_f16(a1, kb0, aK[1][0]);
    aK[1][1] = wmma_f16(a1, kb1, aK[1][1]);
    aV[0][0] = wmma_f16(a0, vb0, aV[0][0]);
    aV[0][1] = wmma_f16(a0, vb1, aV[0][1]);
    aV[1][0] = wmma_f16(a1, vb0, aV[1][0]);
    aV[1][1] = wmma_f16(a1, vb1, aV[1][1]);
  }
  #pragma unroll
  for (int mi = 0; mi < 2; ++mi)
    #pragma unroll
    for (int ni = 0; ni < 2; ++ni) {
      int n = n0 + ni * 16 + (lane & 15);
      int h = n >> 6, c = n & 63;
      float bkn = bk[n], bvn = bv[n];
      #pragma unroll
      for (int r = 0; r < 8; ++r) {
        int m = m0 + mi * 16 + ((lane >> 4) << 3) + r;   // token index
        int b = m >> 12, l = m & (L_ - 1);
        size_t o = (((size_t)(b * H_ + h) * L_ + l) * DH_) + c;
        kh[o] = (_Float16)(aK[mi][ni][r] + bkn);
        vh[o] = (_Float16)(aV[mi][ni][r] + bvn);
      }
    }
}

// ---------------------------------------------------------------------------
// Q projection GEMM: q[g,n] = sum_k tt[g,k]*Wq[n,k] + bq[n]; out f16 [H,G,DH]
// ---------------------------------------------------------------------------
__global__ void __launch_bounds__(256)
q_gemm_kernel(const _Float16* __restrict__ tt, const _Float16* __restrict__ wq,
              const float* __restrict__ bq, _Float16* __restrict__ qh) {
  const int lane = threadIdx.x & 31;
  const int wave = threadIdx.x >> 5;
  const int m0 = blockIdx.y * 128 + (wave >> 1) * 32;
  const int n0 = blockIdx.x * 64  + (wave & 1) * 32;

  v8f acc[2][2] = {};
  for (int k0 = 0; k0 < D_; k0 += 32) {
    v16h a0 = load_a_frag(tt, D_, m0,      k0, lane);
    v16h a1 = load_a_frag(tt, D_, m0 + 16, k0, lane);
    v16h b0 = load_bt_frag(wq, D_, n0,      k0, lane);
    v16h b1 = load_bt_frag(wq, D_, n0 + 16, k0, lane);
    acc[0][0] = wmma_f16(a0, b0, acc[0][0]);
    acc[0][1] = wmma_f16(a0, b1, acc[0][1]);
    acc[1][0] = wmma_f16(a1, b0, acc[1][0]);
    acc[1][1] = wmma_f16(a1, b1, acc[1][1]);
  }
  #pragma unroll
  for (int mi = 0; mi < 2; ++mi)
    #pragma unroll
    for (int ni = 0; ni < 2; ++ni) {
      int n = n0 + ni * 16 + (lane & 15);
      int h = n >> 6, c = n & 63;
      float bn = bq[n];
      #pragma unroll
      for (int r = 0; r < 8; ++r) {
        int g = m0 + mi * 16 + ((lane >> 4) << 3) + r;
        qh[((size_t)h * G_ + g) * DH_ + c] = (_Float16)(acc[mi][ni][r] + bn);
      }
    }
}

// ---------------------------------------------------------------------------
// Fused attention: scores (WMMA) -> top-64 (per-wave argmax) -> mean(v) pool.
// Block = 512 threads (16 waves) handles one (b, h, 16-row g-tile).
// Dynamic LDS: scores[16][4096] f32 = 256 KB (within the 320 KB WGP budget).
// Softmax is skipped: it is monotonic, top-k indices are unchanged.
// ---------------------------------------------------------------------------
__global__ void __launch_bounds__(512)
attn_topk_kernel(const _Float16* __restrict__ qh, const _Float16* __restrict__ kh,
                 const _Float16* __restrict__ vh, float* __restrict__ xo,
                 unsigned* __restrict__ mm) {
  extern __shared__ float smem[];
  float (*S)[L_] = (float(*)[L_])smem;

  const int lane = threadIdx.x & 31;
  const int wave = threadIdx.x >> 5;           // 0..15
  const int gt = blockIdx.x & 31;              // G/16 tiles
  const int bh = blockIdx.x >> 5;              // b*H + h
  const int b  = bh >> 3, h = bh & 7;
  const int g0 = gt * 16;

  const _Float16* qbase = qh + ((size_t)h * G_ + g0) * DH_;
  const _Float16* kbase = kh + (size_t)bh * L_ * DH_;
  const _Float16* vbase = vh + (size_t)bh * L_ * DH_;

  // q fragments live in registers for all 256 l-tiles
  v16h qa0 = load_a_frag(qbase, DH_, 0, 0,  lane);
  v16h qa1 = load_a_frag(qbase, DH_, 0, 32, lane);

  for (int t = wave; t < L_ / 16; t += 16) {
    int l0 = t * 16;
    v16h kb0 = load_bt_frag(kbase, DH_, l0, 0,  lane);
    v16h kb1 = load_bt_frag(kbase, DH_, l0, 32, lane);
    v8f c = {};
    c = wmma_f16(qa0, kb0, c);
    c = wmma_f16(qa1, kb1, c);
    int n = l0 + (lane & 15);
    int rbase = (lane >> 4) << 3;
    #pragma unroll
    for (int r = 0; r < 8; ++r) S[rbase + r][n] = c[r] * SCALE_;
  }
  __syncthreads();

  // one wave per g-row: 64 x (argmax over 4096, invalidate, gather v-row)
  const int row = wave;
  float acc0 = 0.f, acc1 = 0.f;
  for (int it = 0; it < K_; ++it) {
    float best = -3.4e38f;
    int bidx = 0x7FFFFFFF;
    for (int j = lane; j < L_; j += 32) {
      float s = S[row][j];
      if (s > best) { best = s; bidx = j; }
    }
    #pragma unroll
    for (int off = 16; off; off >>= 1) {
      float ov = __shfl_xor(best, off, 32);
      int   oi = __shfl_xor(bidx, off, 32);
      if (ov > best || (ov == best && oi < bidx)) { best = ov; bidx = oi; }
    }
    if (lane == 0) S[row][bidx] = -3.4e38f;   // same wave: LDS ops are in-order
    const _Float16* vr = vbase + (size_t)bidx * DH_ + lane * 2;
    acc0 += (float)vr[0];
    acc1 += (float)vr[1];
  }
  acc0 *= (1.f / K_);
  acc1 *= (1.f / K_);

  size_t o = (((size_t)bh * G_) + g0 + row) * DH_ + lane * 2;
  xo[o]     = acc0;
  xo[o + 1] = acc1;

  float mn = fminf(acc0, acc1), mx = fmaxf(acc0, acc1);
  #pragma unroll
  for (int off = 16; off; off >>= 1) {
    mn = fminf(mn, __shfl_xor(mn, off, 32));
    mx = fmaxf(mx, __shfl_xor(mx, off, 32));
  }
  if (lane == 0) {
    atomicMin(&mm[0], enc_f32(mn));
    atomicMax(&mm[1], enc_f32(mx));
  }
}

// ---------------------------------------------------------------------------
// Global min-max normalize + exp + repack [b,h,g,c] -> f16 [b*G + g, h*64 + c]
// ---------------------------------------------------------------------------
__global__ void normalize_kernel(const float* __restrict__ xo,
                                 const unsigned* __restrict__ mm,
                                 _Float16* __restrict__ ah) {
  int i = blockIdx.x * blockDim.x + threadIdx.x;   // over B*H*G*DH
  float mn = dec_f32(mm[0]);
  float mx = dec_f32(mm[1]);
  float inv = 1.f / (mx - mn);
  int c = i & 63;
  int g = (i >> 6) & (G_ - 1);
  int bh = i >> 15;
  int b = bh >> 3, h = bh & 7;
  float e = expf((xo[i] - mn) * inv);
  ah[((size_t)b * G_ + g) * D_ + h * DH_ + c] = (_Float16)e;
}

// ---------------------------------------------------------------------------
// Output GEMM: out[m,n] = sum_k ah[m,k]*Wout[n,k] + bout[n], f32 out [B,G,D]
// ---------------------------------------------------------------------------
__global__ void __launch_bounds__(256)
out_gemm_kernel(const _Float16* __restrict__ ah, const _Float16* __restrict__ wo,
                const float* __restrict__ bo, float* __restrict__ out) {
  const int lane = threadIdx.x & 31;
  const int wave = threadIdx.x >> 5;
  const int m0 = blockIdx.y * 128 + (wave >> 1) * 32;
  const int n0 = blockIdx.x * 64  + (wave & 1) * 32;

  v8f acc[2][2] = {};
  for (int k0 = 0; k0 < D_; k0 += 32) {
    v16h a0 = load_a_frag(ah, D_, m0,      k0, lane);
    v16h a1 = load_a_frag(ah, D_, m0 + 16, k0, lane);
    v16h b0 = load_bt_frag(wo, D_, n0,      k0, lane);
    v16h b1 = load_bt_frag(wo, D_, n0 + 16, k0, lane);
    acc[0][0] = wmma_f16(a0, b0, acc[0][0]);
    acc[0][1] = wmma_f16(a0, b1, acc[0][1]);
    acc[1][0] = wmma_f16(a1, b0, acc[1][0]);
    acc[1][1] = wmma_f16(a1, b1, acc[1][1]);
  }
  #pragma unroll
  for (int mi = 0; mi < 2; ++mi)
    #pragma unroll
    for (int ni = 0; ni < 2; ++ni) {
      int n = n0 + ni * 16 + (lane & 15);
      float bn = bo[n];
      #pragma unroll
      for (int r = 0; r < 8; ++r) {
        int m = m0 + mi * 16 + ((lane >> 4) << 3) + r;
        out[(size_t)m * D_ + n] = acc[mi][ni][r] + bn;
      }
    }
}

// ---------------------------------------------------------------------------
// Host launcher
// ---------------------------------------------------------------------------
extern "C" void kernel_launch(void* const* d_in, const int* in_sizes, int n_in,
                              void* d_out, int out_size, void* d_ws, size_t ws_size,
                              hipStream_t stream) {
  (void)in_sizes; (void)n_in; (void)out_size; (void)ws_size;
  const float* x   = (const float*)d_in[0];
  const float* tt  = (const float*)d_in[1];
  const float* Wq  = (const float*)d_in[2];
  const float* bq  = (const float*)d_in[3];
  const float* Wk  = (const float*)d_in[4];
  const float* bk  = (const float*)d_in[5];
  const float* Wv  = (const float*)d_in[6];
  const float* bv  = (const float*)d_in[7];
  const float* Wo  = (const float*)d_in[8];
  const float* bo  = (const float*)d_in[9];

  char* ws = (char*)d_ws;
  _Float16* xh  = (_Float16*)(ws + OFF_XH);
  _Float16* kh  = (_Float16*)(ws + OFF_KH);
  _Float16* vh  = (_Float16*)(ws + OFF_VH);
  _Float16* qh  = (_Float16*)(ws + OFF_QH);
  _Float16* wqh = (_Float16*)(ws + OFF_WQ);
  _Float16* wkh = (_Float16*)(ws + OFF_WK);
  _Float16* wvh = (_Float16*)(ws + OFF_WV);
  _Float16* woh = (_Float16*)(ws + OFF_WO);
  _Float16* tth = (_Float16*)(ws + OFF_TT);
  float*    xo  = (float*)(ws + OFF_XO);
  _Float16* ah  = (_Float16*)(ws + OFF_AH);
  unsigned* mm  = (unsigned*)(ws + OFF_MM);

  // --- prep: f16 conversions (+ fused log1p(relu) for x) ---
  {
    int n4 = (BL_ * D_) / 4;
    log1p_f16_kernel<<<(n4 + 255) / 256, 256, 0, stream>>>(x, xh, n4);
    int w4 = (D_ * D_) / 4;
    cvt_f16_kernel<<<(w4 + 255) / 256, 256, 0, stream>>>(Wq, wqh, w4);
    cvt_f16_kernel<<<(w4 + 255) / 256, 256, 0, stream>>>(Wk, wkh, w4);
    cvt_f16_kernel<<<(w4 + 255) / 256, 256, 0, stream>>>(Wv, wvh, w4);
    cvt_f16_kernel<<<(w4 + 255) / 256, 256, 0, stream>>>(Wo, woh, w4);
    int t4 = (G_ * D_) / 4;
    cvt_f16_kernel<<<(t4 + 255) / 256, 256, 0, stream>>>(tt, tth, t4);
    init_minmax_kernel<<<1, 1, 0, stream>>>(mm);
  }

  // --- projections (WMMA) ---
  q_gemm_kernel<<<dim3(D_ / 64, G_ / 128), 256, 0, stream>>>(tth, wqh, bq, qh);
  kv_gemm_kernel<<<dim3(D_ / 64, BL_ / 128), 256, 0, stream>>>(xh, wkh, wvh, bk, bv, kh, vh);

  // --- fused scores + top-k + mean pool (256 KB dynamic LDS) ---
  {
    size_t shmem = (size_t)16 * L_ * sizeof(float);   // 262144 B
    (void)hipFuncSetAttribute((const void*)attn_topk_kernel,
                              hipFuncAttributeMaxDynamicSharedMemorySize,
                              (int)shmem);
    int grid = B_ * H_ * (G_ / 16);                   // 2048
    attn_topk_kernel<<<grid, 512, shmem, stream>>>(qh, kh, vh, xo, mm);
  }

  // --- epilogue normalize + output projection (WMMA) ---
  {
    int n = B_ * H_ * G_ * DH_;
    normalize_kernel<<<n / 256, 256, 0, stream>>>(xo, mm, ah);
    out_gemm_kernel<<<dim3(D_ / 64, (B_ * G_) / 128), 256, 0, stream>>>(
        ah, woh, bo, (float*)d_out);
  }
}
